// QuantFlow_82952998355346
// MI455X (gfx1250) — compile-verified, matrix operand order checked by reading, not run on hardware
//
#include <hip/hip_runtime.h>
#include <hip/hip_bf16.h>
#include <math.h>

// ---- problem constants (from reference) ----
#define Bsz 8
#define SEQ 512
#define NV 128
#define DM 512
#define DSTATE 16
#define DFF 2048
#define DIN 1024
#define DTR 32
#define PRED 96
#define NQ 3
#define NT (Bsz*NV)   // 1024 tokens after inverted embedding

typedef _Float16 v16h __attribute__((ext_vector_type(16)));
typedef _Float16 h8   __attribute__((ext_vector_type(8)));
typedef _Float16 h4   __attribute__((ext_vector_type(4)));
typedef float    v8f  __attribute__((ext_vector_type(8)));

enum { EP_NONE=0, EP_BIAS=1, EP_BIAS_GELU=2, EP_BIAS_SOFTPLUS=3 };

__device__ __forceinline__ float siluf(float x){ return x / (1.f + __expf(-x)); }

// =====================================================================
// Generic WMMA GEMM:  C[M,N] = A[M,K] * B[K,N] (+epilogue)
//   A(m,k) at A + z*strideAb + m*ldaM + k*ldaK   (fp32 -> f16 in LDS)
//   B(k,n) at B + n*ldbN + k                     (weights [N,K] row-major)
//   C(m,n) at C + z*strideCb + m*ldcM + n        (fp32)
// Assumptions (hold for every call site): M % 64 == 0, K % 32 == 0,
//   B is K-contiguous; A is K-contiguous (aKcontig=1) or M-contiguous
//   (aKcontig=0, ldaM==1, used by the inverted-embedding GEMM).
// Block: 128 threads = 4 waves; tile 64x64; each wave owns 32x32 (4 WMMAs).
// =====================================================================
__global__ __launch_bounds__(128) void wmma_gemm_kernel(
    const float* __restrict__ A, const float* __restrict__ B,
    const float* __restrict__ bias, float* __restrict__ C,
    int M, int N, int K,
    long long ldaM, long long ldaK, long long strideAb,
    long long ldbN,
    long long ldcM, long long strideCb, int ep, int aKcontig)
{
    // [row][k] layout for BOTH operands; row stride 40 halves = 80 B = 5*16 B
    __shared__ _Float16 As[64][40];
    __shared__ _Float16 Bs[64][40];

    const int tid  = threadIdx.x;
    const int lane = tid & 31;
    const int wave = tid >> 5;
    const int m0 = blockIdx.y * 64;
    const int n0 = blockIdx.x * 64;
    const float* Ab = A + (long long)blockIdx.z * strideAb;
    float*       Cb = C + (long long)blockIdx.z * strideCb;

    const int wm = (wave >> 1) * 32;   // wave tile origin inside block tile
    const int wn = (wave & 1)  * 32;

    v8f acc[2][2];
    #pragma unroll
    for (int mi=0;mi<2;++mi)
      #pragma unroll
      for (int ni=0;ni<2;++ni)
        #pragma unroll
        for (int r=0;r<8;++r) acc[mi][ni][r] = 0.f;

    // wave32 f16 fragment layout (ISA 7.12.2): per lane, element j=0..7 is
    // K = hi8+0..7 and j=8..15 is K = hi8+16..23, hi8 = 8*(lane>=16).
    const int hi8  = ((lane >> 4) & 1) * 8;
    const int mrow = lane & 15;

    for (int k0 = 0; k0 < K; k0 += 32){
        // ---- stage A tile (64 rows x 32 k), vectorized ----
        if (aKcontig){
            #pragma unroll
            for (int it=0; it<4; ++it){
                int idx = tid + it*128;           // 512 float4 loads total
                int r = idx >> 3;                 // row 0..63
                int c = (idx & 7) << 2;           // k   0..28 step 4
                float4 v = *(const float4*)(Ab + (long long)(m0+r)*ldaM + (k0+c));
                h4 hv; hv[0]=(_Float16)v.x; hv[1]=(_Float16)v.y;
                       hv[2]=(_Float16)v.z; hv[3]=(_Float16)v.w;
                *(h4*)&As[r][c] = hv;
            }
        } else {
            // m-contiguous (ldaM == 1): vector load along m, scatter k column
            #pragma unroll
            for (int it=0; it<4; ++it){
                int idx = tid + it*128;
                int c = idx >> 4;                 // k 0..31
                int r = (idx & 15) << 2;          // m 0..60 step 4
                float4 v = *(const float4*)(Ab + (long long)(m0+r) + (long long)(k0+c)*ldaK);
                As[r+0][c]=(_Float16)v.x; As[r+1][c]=(_Float16)v.y;
                As[r+2][c]=(_Float16)v.z; As[r+3][c]=(_Float16)v.w;
            }
        }
        // ---- stage B tile (64 n-rows x 32 k), row-level N guard ----
        #pragma unroll
        for (int it=0; it<4; ++it){
            int idx = tid + it*128;
            int r = idx >> 3;                     // n row 0..63
            int c = (idx & 7) << 2;               // k   0..28 step 4
            float4 v = make_float4(0.f,0.f,0.f,0.f);
            if (n0 + r < N) v = *(const float4*)(B + (long long)(n0+r)*ldbN + (k0+c));
            h4 hv; hv[0]=(_Float16)v.x; hv[1]=(_Float16)v.y;
                   hv[2]=(_Float16)v.z; hv[3]=(_Float16)v.w;
            *(h4*)&Bs[r][c] = hv;
        }
        // hint next weight tile toward L2 (global_prefetch_b8)
        if (k0 + 32 < K){
            int pn = n0 + (tid & 63);
            if (pn < N) __builtin_prefetch(B + (long long)pn*ldbN + (k0+32), 0, 1);
        }
        __syncthreads();

        #pragma unroll
        for (int mi=0; mi<2; ++mi){
            const _Float16* ap = &As[wm + mi*16 + mrow][0];
            h8 alo = *(const h8*)(ap + hi8);
            h8 ahi = *(const h8*)(ap + hi8 + 16);
            v16h a = __builtin_shufflevector(alo, ahi,
                        0,1,2,3,4,5,6,7,8,9,10,11,12,13,14,15);
            #pragma unroll
            for (int ni=0; ni<2; ++ni){
                const _Float16* bp = &Bs[wn + ni*16 + mrow][0];
                h8 blo = *(const h8*)(bp + hi8);
                h8 bhi = *(const h8*)(bp + hi8 + 16);
                v16h b = __builtin_shufflevector(blo, bhi,
                            0,1,2,3,4,5,6,7,8,9,10,11,12,13,14,15);
                acc[mi][ni] = __builtin_amdgcn_wmma_f32_16x16x32_f16(
                    false, a, false, b, (short)0, acc[mi][ni], false, false);
            }
        }
        __syncthreads();
    }

    // C/D layout: N = lane&15, M = vgpr + 8*(lane>=16)
    const int nl = lane & 15;
    const int mh = (lane >> 4) * 8;
    #pragma unroll
    for (int mi=0; mi<2; ++mi)
      #pragma unroll
      for (int ni=0; ni<2; ++ni)
        #pragma unroll
        for (int r=0; r<8; ++r){
            int gm = m0 + wm + mi*16 + mh + r;
            int gn = n0 + wn + ni*16 + nl;
            if (gn < N){
                float x = acc[mi][ni][r];
                if (ep != EP_NONE) x += bias[gn];
                if (ep == EP_BIAS_GELU)          x = 0.5f * x * (1.f + erff(x * 0.70710678118654752f));
                else if (ep == EP_BIAS_SOFTPLUS) x = (x > 20.f) ? x : log1pf(__expf(x));
                Cb[(long long)gm*ldcM + gn] = x;
            }
        }
}

// ---- instance-norm stats over time per (b, variate) ----
__global__ __launch_bounds__(128) void stats_kernel(
    const float* __restrict__ x, float* __restrict__ mean, float* __restrict__ stdv)
{
    __shared__ float r1[128], r2[128];
    const int bv = blockIdx.x, b = bv / NV, v = bv % NV;
    float s = 0.f, s2 = 0.f;
    for (int i = threadIdx.x; i < SEQ; i += 128){
        float val = x[((long long)b*SEQ + i)*NV + v];
        s += val; s2 += val*val;
    }
    r1[threadIdx.x] = s; r2[threadIdx.x] = s2; __syncthreads();
    for (int off=64; off>0; off>>=1){
        if (threadIdx.x < off){ r1[threadIdx.x]+=r1[threadIdx.x+off]; r2[threadIdx.x]+=r2[threadIdx.x+off]; }
        __syncthreads();
    }
    if (threadIdx.x == 0){
        float mu  = r1[0] * (1.f/SEQ);
        float var = r2[0] * (1.f/SEQ) - mu*mu;
        mean[bv] = mu;
        stdv[bv] = sqrtf(var + 1e-5f);
    }
}

__global__ void xn_kernel(const float* __restrict__ x, const float* __restrict__ mean,
                          const float* __restrict__ stdv, float* __restrict__ xn)
{
    int g = blockIdx.x*256 + threadIdx.x;
    if (g >= Bsz*SEQ*NV) return;
    int v = g % NV;
    int b = g / (SEQ*NV);
    xn[g] = (x[g] - mean[b*NV+v]) / stdv[b*NV+v];
}

__global__ void flip_kernel(const float* __restrict__ src, float* __restrict__ dst)
{
    int g = blockIdx.x*256 + threadIdx.x;
    if (g >= NT*DM) return;
    int d = g % DM, t = g / DM, b = t / NV, l = t % NV;
    dst[((long long)(b*NV + (NV-1-l)))*DM + d] = src[g];
}

// causal depthwise conv(k=4) over variate axis + SiLU; reads x half of xz
__global__ void conv_silu_kernel(const float* __restrict__ xz, const float* __restrict__ cw,
                                 const float* __restrict__ cb, float* __restrict__ xc)
{
    int g = blockIdx.x*256 + threadIdx.x;
    if (g >= NT*DIN) return;
    int d = g % DIN, t = g / DIN, b = t / NV, l = t % NV;
    float s = cb[d];
    #pragma unroll
    for (int k=0;k<4;++k){
        int ls = l - 3 + k;
        if (ls >= 0) s += cw[d*4+k] * xz[((long long)(b*NV + ls))*(2*DIN) + d];
    }
    xc[g] = siluf(s);
}

// selective scan: one thread per (b, d_inner); 128 steps x 16 states
__global__ void scan_kernel(const float* __restrict__ xc, const float* __restrict__ dt,
                            const float* __restrict__ xdbl, const float* __restrict__ A_log,
                            const float* __restrict__ Dp, float* __restrict__ y)
{
    int g = blockIdx.x*256 + threadIdx.x;
    if (g >= Bsz*DIN) return;
    int b = g / DIN, d = g % DIN;
    float Arow[DSTATE], h[DSTATE];
    #pragma unroll
    for (int n=0;n<DSTATE;++n){ Arow[n] = -__expf(A_log[d*DSTATE + n]); h[n] = 0.f; }
    const float Dd = Dp[d];
    for (int l=0; l<NV; ++l){
        long long t = (long long)(b*NV + l);
        float u   = xc[t*DIN + d];
        float dtv = dt[t*DIN + d];
        const float* xr = xdbl + t*(DTR + 2*DSTATE);
        float acc = 0.f;
        #pragma unroll
        for (int n=0;n<DSTATE;++n){
            float dA = __expf(dtv * Arow[n]);
            h[n] = dA * h[n] + dtv * xr[DTR + n] * u;
            acc += h[n] * xr[DTR + DSTATE + n];
        }
        y[t*DIN + d] = acc + u * Dd;
    }
}

__global__ void gate_kernel(const float* __restrict__ scy, const float* __restrict__ xz,
                            float* __restrict__ gat)
{
    int g = blockIdx.x*256 + threadIdx.x;
    if (g >= NT*DIN) return;
    int d = g % DIN; long long t = g / DIN;
    float z = xz[t*(2*DIN) + DIN + d];
    gat[g] = scy[g] * siluf(z);
}

// out = LN(base + a1 + flip?(a2)) ; a1/a2 optional
__global__ __launch_bounds__(128) void addln_kernel(
    float* __restrict__ out, const float* __restrict__ base,
    const float* __restrict__ a1, const float* __restrict__ a2, int flip2,
    const float* __restrict__ w, const float* __restrict__ bb)
{
    __shared__ float red[128];
    const int t = blockIdx.x, tid = threadIdx.x;
    long long r = (long long)t * DM, r2 = r;
    if (flip2){ int b = t / NV, l = t % NV; r2 = (long long)(b*NV + (NV-1-l)) * DM; }
    float vals[4]; float s = 0.f;
    #pragma unroll
    for (int i=0;i<4;++i){
        int c = tid + i*128;
        float v = base[r+c];
        if (a1) v += a1[r+c];
        if (a2) v += a2[r2+c];
        vals[i] = v; s += v;
    }
    red[tid] = s; __syncthreads();
    for (int off=64; off>0; off>>=1){ if (tid<off) red[tid]+=red[tid+off]; __syncthreads(); }
    float mu = red[0] * (1.f/DM); __syncthreads();
    float s2 = 0.f;
    #pragma unroll
    for (int i=0;i<4;++i){ float d = vals[i]-mu; s2 += d*d; }
    red[tid] = s2; __syncthreads();
    for (int off=64; off>0; off>>=1){ if (tid<off) red[tid]+=red[tid+off]; __syncthreads(); }
    float rstd = rsqrtf(red[0] * (1.f/DM) + 1e-5f);
    #pragma unroll
    for (int i=0;i<4;++i){
        int c = tid + i*128;
        out[r+c] = (vals[i]-mu)*rstd*w[c] + bb[c];
    }
}

// de-normalize + [B,V,PRED*NQ] -> [B,PRED,V,NQ]
__global__ void out_kernel(const float* __restrict__ dec, const float* __restrict__ mean,
                           const float* __restrict__ stdv, float* __restrict__ out)
{
    int g = blockIdx.x*256 + threadIdx.x;
    if (g >= Bsz*PRED*NV*NQ) return;
    int q = g % NQ;
    int v = (g / NQ) % NV;
    int p = (g / (NQ*NV)) % PRED;
    int b = g / (NQ*NV*PRED);
    float val = dec[((long long)(b*NV + v))*(PRED*NQ) + p*NQ + q];
    out[g] = val * stdv[b*NV+v] + mean[b*NV+v];
}

// =====================================================================
// host-side orchestration
// =====================================================================
static void gemm(hipStream_t st, const float* A, const float* B, const float* bias, float* C,
                 int M, int N, int K, long long ldaM, long long ldaK, long long sAb,
                 long long ldbN, long long ldcM, long long sCb,
                 int ep, int zb)
{
    dim3 grid((N+63)/64, (M+63)/64, zb);
    int aKcontig = (ldaK == 1) ? 1 : 0;
    wmma_gemm_kernel<<<grid, 128, 0, st>>>(A, B, bias, C, M, N, K,
                                           ldaM, ldaK, sAb, ldbN, ldcM, sCb, ep, aKcontig);
}

struct MambaP {
    const float *in_proj, *conv_w, *conv_b, *x_proj, *dt_w, *dt_b, *A_log, *D, *out_proj;
};
static MambaP get_mamba(void* const* d_in, int base){
    MambaP p;
    p.in_proj  = (const float*)d_in[base+0];
    p.conv_w   = (const float*)d_in[base+1];
    p.conv_b   = (const float*)d_in[base+2];
    p.x_proj   = (const float*)d_in[base+3];
    p.dt_w     = (const float*)d_in[base+4];
    p.dt_b     = (const float*)d_in[base+5];
    p.A_log    = (const float*)d_in[base+6];
    p.D        = (const float*)d_in[base+7];
    p.out_proj = (const float*)d_in[base+8];
    return p;
}

static void run_mamba(hipStream_t st, const float* x, const MambaP& p, float* out,
                      float* xz, float* xc, float* xdbl, float* dtb, float* scy, float* gat)
{
    // xz = x @ in_proj.T                              [NT, 2048]
    gemm(st, x, p.in_proj, nullptr, xz, NT, 2*DIN, DM, DM,1,0, DM, 2*DIN,0, EP_NONE, 1);
    conv_silu_kernel<<<(NT*DIN)/256, 256, 0, st>>>(xz, p.conv_w, p.conv_b, xc);
    // xdbl = xc @ x_proj.T                            [NT, 64]
    gemm(st, xc, p.x_proj, nullptr, xdbl, NT, DTR+2*DSTATE, DIN, DIN,1,0, DIN, DTR+2*DSTATE,0, EP_NONE, 1);
    // dt = softplus(xdbl[:, :32] @ dt_w.T + dt_b)     [NT, 1024]
    gemm(st, xdbl, p.dt_w, p.dt_b, dtb, NT, DIN, DTR, DTR+2*DSTATE,1,0, DTR, DIN,0, EP_BIAS_SOFTPLUS, 1);
    scan_kernel<<<(Bsz*DIN)/256, 256, 0, st>>>(xc, dtb, xdbl, p.A_log, p.D, scy);
    gate_kernel<<<(NT*DIN)/256, 256, 0, st>>>(scy, xz, gat);
    // out = gated @ out_proj.T                        [NT, 512]
    gemm(st, gat, p.out_proj, nullptr, out, NT, DM, DIN, DIN,1,0, DIN, DM,0, EP_NONE, 1);
}

extern "C" void kernel_launch(void* const* d_in, const int* in_sizes, int n_in,
                              void* d_out, int out_size, void* d_ws, size_t ws_size,
                              hipStream_t stream)
{
    (void)in_sizes; (void)n_in; (void)out_size; (void)ws_size;
    const float* xin    = (const float*)d_in[0];
    const float* emb_w  = (const float*)d_in[1];
    const float* emb_b  = (const float*)d_in[2];
    const float* norm_w = (const float*)d_in[55];
    const float* norm_b = (const float*)d_in[56];
    const float* proj_w = (const float*)d_in[57];
    const float* proj_b = (const float*)d_in[58];

    float* W = (float*)d_ws;
    size_t o = 0;
    float* meanp = W+o; o += Bsz*NV;
    float* stdp  = W+o; o += Bsz*NV;
    float* xn    = W+o; o += (size_t)Bsz*SEQ*NV;
    float* enc   = W+o; o += (size_t)NT*DM;
    float* encF  = W+o; o += (size_t)NT*DM;
    float* fwd   = W+o; o += (size_t)NT*DM;     // also reused as FFN output
    float* bwd   = W+o; o += (size_t)NT*DM;
    float* xz    = W+o; o += (size_t)NT*2*DIN;
    float* xc    = W+o; o += (size_t)NT*DIN;
    float* xdbl  = W+o; o += (size_t)NT*(DTR+2*DSTATE);
    float* dtb   = W+o; o += (size_t)NT*DIN;
    float* scy   = W+o; o += (size_t)NT*DIN;
    float* gat   = W+o; o += (size_t)NT*DIN;
    float* ff1   = W+o; o += (size_t)NT*DFF;
    float* dec   = W+o; o += (size_t)NT*PRED*NQ;

    // instance norm over time + inverted embedding GEMM
    stats_kernel<<<Bsz*NV, 128, 0, stream>>>(xin, meanp, stdp);
    xn_kernel<<<(Bsz*SEQ*NV)/256, 256, 0, stream>>>(xin, meanp, stdp, xn);
    // enc[b,v,:] = xn[b,:,v] @ emb_w.T + emb_b   (A strided: ldaM=1, ldaK=NV)
    gemm(stream, xn, emb_w, emb_b, enc, NV, DM, SEQ,
         1, NV, (long long)SEQ*NV, SEQ, DM, (long long)NV*DM, EP_BIAS, Bsz);

    for (int L = 0; L < 2; ++L){
        int base = 3 + L*26;
        MambaP pf = get_mamba(d_in, base + 0);
        MambaP pb = get_mamba(d_in, base + 9);
        const float* w1   = (const float*)d_in[base+18];
        const float* b1   = (const float*)d_in[base+19];
        const float* w2   = (const float*)d_in[base+20];
        const float* b2   = (const float*)d_in[base+21];
        const float* ln1w = (const float*)d_in[base+22];
        const float* ln1b = (const float*)d_in[base+23];
        const float* ln2w = (const float*)d_in[base+24];
        const float* ln2b = (const float*)d_in[base+25];

        flip_kernel<<<(NT*DM)/256, 256, 0, stream>>>(enc, encF);
        run_mamba(stream, enc,  pf, fwd, xz, xc, xdbl, dtb, scy, gat);
        run_mamba(stream, encF, pb, bwd, xz, xc, xdbl, dtb, scy, gat);
        // enc = LN(enc + fwd + flip(bwd))
        addln_kernel<<<NT, 128, 0, stream>>>(enc, enc, fwd, bwd, 1, ln1w, ln1b);
        // FFN: gelu(enc@w1.T+b1) @ w2.T + b2
        gemm(stream, enc, w1, b1, ff1, NT, DFF, DM, DM,1,0, DM, DFF,0, EP_BIAS_GELU, 1);
        gemm(stream, ff1, w2, b2, fwd, NT, DM, DFF, DFF,1,0, DFF, DM,0, EP_BIAS, 1);
        addln_kernel<<<NT, 128, 0, stream>>>(enc, enc, fwd, nullptr, 0, ln2w, ln2b);
    }

    // final LN + projection + de-normalize/transpose
    addln_kernel<<<NT, 128, 0, stream>>>(enc, enc, nullptr, nullptr, 0, norm_w, norm_b);
    gemm(stream, enc, proj_w, proj_b, dec, NT, PRED*NQ, DM, DM,1,0, DM, PRED*NQ,0, EP_BIAS, 1);
    out_kernel<<<(Bsz*PRED*NV*NQ)/256, 256, 0, stream>>>(dec, meanp, stdp, (float*)d_out);
}
